// SwarmNet_58540404244716
// MI455X (gfx1250) — compile-verified, matrix operand order checked by reading, not run on hardware
//
#include <hip/hip_runtime.h>
#include <math.h>

// ---------------------------------------------------------------------------
// SwarmNet one-step predictor for MI455X (gfx1250, wave32, WMMA).
//
// Dominant cost: agg = A_nd @ cond  ->  4096x4096 (0/1 adjacency) x 4096x256
// (4 batches x 64 features fused). 64 MB adjacency read @ 23.3 TB/s ~ 2.7us;
// we use v_wmma_f32_16x16x32_f16 with a hi/lo f16 split of cond (A is exactly
// representable in f16) so products are exact in the f32 accumulator ->
// ~f32 accuracy at f16 matrix throughput. deg / col_deg / diagonal correction
// are fused into the same single pass over A.
// ---------------------------------------------------------------------------

typedef _Float16 v16h __attribute__((ext_vector_type(16)));
typedef float    v8f  __attribute__((ext_vector_type(8)));

#define NN  4096
#define BB  4
#define DD  4
#define TT  5
#define F1C 32
#define F2C 64
#define BF  256   // BB * F2C fused output columns
#define MT  64    // output rows per workgroup in the GEMM
#define KT  32    // K depth per iteration (= one WMMA K)

// WMMA 16x16x32 f16 operand swizzle (A: p=row-in-16, B: p=col-in-16):
//   lane = p + 16*((k%16)/8) ; elem = (k/16)*8 + (k%8)
__device__ __forceinline__ int frag_lane(int p, int k) { return p + 16 * ((k & 15) >> 3); }
__device__ __forceinline__ int frag_elem(int k)        { return ((k >> 4) << 3) + (k & 7); }

// ---------------------------------------------------------------------------
__global__ __launch_bounds__(256) void zero_kernel(int* __restrict__ p, int n) {
    int i = blockIdx.x * 256 + threadIdx.x;
    if (i < n) p[i] = 0;
}

// ---------------------------------------------------------------------------
// Kernel 1: temporal convs -> cond (f32) + cond hi/lo (f16) pre-swizzled into
// WMMA B-fragment order so the GEMM stages LDS with contiguous uint4 copies.
// One thread per (b, n).
// ---------------------------------------------------------------------------
__global__ __launch_bounds__(256) void cond_kernel(
    const float* __restrict__ ts,
    const float* __restrict__ Wc1, const float* __restrict__ bc1,
    const float* __restrict__ Wc2, const float* __restrict__ bc2,
    float* __restrict__ cond_f32,
    _Float16* __restrict__ cond_hi, _Float16* __restrict__ cond_lo)
{
    __shared__ float w1[3 * DD * F1C];     // 384
    __shared__ float w2[3 * F1C * F2C];    // 6144
    __shared__ float b1[F1C], b2[F2C];
    const int t = threadIdx.x;
    for (int i = t; i < 3 * DD * F1C; i += 256) w1[i] = Wc1[i];
    for (int i = t; i < 3 * F1C * F2C; i += 256) w2[i] = Wc2[i];
    if (t < F1C) b1[t] = bc1[t];
    if (t < F2C) b2[t] = bc2[t];
    __syncthreads();

    const int g = blockIdx.x * 256 + t;
    const int b = g >> 12;
    const int n = g & (NN - 1);

    float x[TT][DD];
    #pragma unroll
    for (int tt = 0; tt < TT; ++tt) {
        const float4 v = *(const float4*)(ts + (((size_t)b * TT + tt) * NN + n) * DD);
        x[tt][0] = v.x; x[tt][1] = v.y; x[tt][2] = v.z; x[tt][3] = v.w;
    }

    // conv1 (k=3, VALID, relu): 5 -> 3 time steps, D -> F1
    float h1[3][F1C];
    #pragma unroll
    for (int to = 0; to < 3; ++to) {
        for (int f = 0; f < F1C; ++f) {
            float s = b1[f];
            #pragma unroll
            for (int kt = 0; kt < 3; ++kt)
                #pragma unroll
                for (int d = 0; d < DD; ++d)
                    s += x[to + kt][d] * w1[(kt * DD + d) * F1C + f];
            h1[to][f] = fmaxf(s, 0.0f);
        }
    }

    // conv2 (k=3, VALID): 3 -> 1 time step, F1 -> F2 (no relu)
    const int kblk = n >> 5, kin = n & 31;
    const int ie = frag_elem(kin);
    for (int f2 = 0; f2 < F2C; ++f2) {
        float s = b2[f2];
        for (int kt = 0; kt < 3; ++kt)
            for (int f = 0; f < F1C; ++f)
                s += h1[kt][f] * w2[(kt * F1C + f) * F2C + f2];   // uniform -> LDS broadcast
        const int bf = b * F2C + f2;
        cond_f32[(size_t)n * BF + bf] = s;
        const _Float16 hi = (_Float16)s;
        const _Float16 lo = (_Float16)(s - (float)hi);
        const int tile = bf >> 4;
        const int lane = frag_lane(bf & 15, kin);
        const size_t addr = (((size_t)kblk * 16 + tile) * 32 + lane) * 16 + ie;
        cond_hi[addr] = hi;
        cond_lo[addr] = lo;
    }
}

// ---------------------------------------------------------------------------
// Kernel 2: fused GEMM  S = A @ [cond_hi | cond_lo]  with row/col degree
// accumulation and the (S - diag*cond) / max(deg,1) epilogue.
// Grid: 64 WGs x 256 threads (8 waves). Each WG: 64 rows x 256 cols, full K.
// Wave tiling: 4 row-tiles x 2 col-halves; 8 accumulators (16x128) per wave.
// ---------------------------------------------------------------------------
__global__ __launch_bounds__(256) void agg_kernel(
    const int* __restrict__ edges,
    const float* __restrict__ cond_f32,
    const _Float16* __restrict__ cond_hi, const _Float16* __restrict__ cond_lo,
    float* __restrict__ agg, int* __restrict__ colDeg)
{
    __shared__ __align__(32) _Float16 As [MT * KT];        // 4 KB, swizzled
    __shared__ __align__(32) _Float16 Bhi[16 * 32 * 16];   // 16 KB, swizzled
    __shared__ __align__(32) _Float16 Blo[16 * 32 * 16];   // 16 KB, swizzled
    __shared__ int colpart[NN];                            // 16 KB
    __shared__ int rowpart[MT];

    const int t     = threadIdx.x;
    const int jbase = blockIdx.x * MT;
    for (int i = t; i < NN; i += 256) colpart[i] = 0;
    if (t < MT) rowpart[t] = 0;

    const int lane = t & 31;
    const int wave = t >> 5;
    const int wm   = wave & 3;   // row tile (16 rows)
    const int wn   = wave >> 2;  // col half (128 cols)

    v8f acc[8] = {};
    int rsum[8];
    #pragma unroll
    for (int i = 0; i < 8; ++i) rsum[i] = 0;

    const int a_r0 = t >> 5;     // thread's base row (rows a_r0 + 8*i)
    const int a_c  = t & 31;     // thread's fixed column within the K tile
    const int a_ie = frag_elem(a_c);
    __syncthreads();

    for (int k0 = 0; k0 < NN; k0 += KT) {
        // --- stage A tile: int 0/1 -> f16, swizzled; accumulate degrees ---
        int csum = 0;
        #pragma unroll
        for (int i = 0; i < 8; ++i) {
            const int r  = a_r0 + 8 * i;
            const int ev = edges[(size_t)(jbase + r) * NN + (k0 + a_c)];
            rsum[i] += ev;
            csum    += ev;
            const int rtile = r >> 4;
            const int al    = frag_lane(r & 15, a_c);
            As[((rtile * 32 + al) << 4) + a_ie] = (_Float16)ev;
        }
        atomicAdd(&colpart[k0 + a_c], csum);

        // --- stage B tiles: contiguous 16 KB copies (pre-swizzled global) ---
        {
            const size_t base = (size_t)(k0 >> 5) * (16 * 32 * 16);
            const uint4* sH = (const uint4*)(cond_hi + base);
            const uint4* sL = (const uint4*)(cond_lo + base);
            uint4* dH = (uint4*)Bhi;
            uint4* dL = (uint4*)Blo;
            #pragma unroll
            for (int i = 0; i < 4; ++i) {
                dH[t + 256 * i] = sH[t + 256 * i];
                dL[t + 256 * i] = sL[t + 256 * i];
            }
        }
        if (k0 + KT < NN)   // stream next A tile into cache (global_prefetch_b8)
            __builtin_prefetch(&edges[(size_t)(jbase + a_r0) * NN + (k0 + KT + a_c)], 0, 1);
        __syncthreads();

        // --- compute: 1 A frag, 8 col tiles x (hi + lo) WMMA ---
        const v16h a = *(const v16h*)&As[(wm * 32 + lane) << 4];
        #pragma unroll
        for (int c = 0; c < 8; ++c) {
            const int tile = wn * 8 + c;
            const v16h bh = *(const v16h*)&Bhi[(tile * 32 + lane) << 4];
            const v16h bl = *(const v16h*)&Blo[(tile * 32 + lane) << 4];
            acc[c] = __builtin_amdgcn_wmma_f32_16x16x32_f16(
                false, a, false, bh, (short)0, acc[c], false, false);
            acc[c] = __builtin_amdgcn_wmma_f32_16x16x32_f16(
                false, a, false, bl, (short)0, acc[c], false, false);
        }
        __syncthreads();
    }

    // --- reduce degrees ---
    #pragma unroll
    for (int i = 0; i < 8; ++i) atomicAdd(&rowpart[a_r0 + 8 * i], rsum[i]);
    __syncthreads();
    for (int i = t; i < NN; i += 256)
        if (colpart[i] != 0) atomicAdd(&colDeg[i], colpart[i]);

    // --- epilogue: agg = (S - diag*cond[j]) / max(deg,1) ---
    const int hh = lane >> 4;
    const int nl = lane & 15;
    #pragma unroll
    for (int vv = 0; vv < 8; ++vv) {
        const int m = wm * 16 + vv + 8 * hh;
        const int j = jbase + m;
        const float scale = 1.0f / fmaxf((float)rowpart[m], 1.0f);
        const float diag  = (float)edges[(size_t)j * NN + j];
        #pragma unroll
        for (int c = 0; c < 8; ++c) {
            const int n = wn * 128 + c * 16 + nl;
            const float cj = cond_f32[(size_t)j * BF + n];
            agg[(size_t)j * BF + n] = (acc[c][vv] - diag * cj) * scale;
        }
    }
}

// ---------------------------------------------------------------------------
// Kernel 3: per-node MLP tail.  One thread per (b, n); weights in LDS so all
// weight reads are wave-uniform LDS broadcasts.
// ---------------------------------------------------------------------------
__global__ __launch_bounds__(256) void out_kernel(
    const float* __restrict__ ts,
    const float* __restrict__ cond_f32, const float* __restrict__ agg,
    const int* __restrict__ colDeg,
    const float* __restrict__ Wg1, const float* __restrict__ bg,
    const float* __restrict__ Wd,  const float* __restrict__ bd,
    const float* __restrict__ Wo,  const float* __restrict__ bo,
    float* __restrict__ out)
{
    __shared__ float wg[64 * 64];
    __shared__ float wd[128 * 64];
    __shared__ float wo[64 * DD];
    __shared__ float sbg[64], sbd[64], sbo[DD];
    const int t = threadIdx.x;
    for (int i = t; i < 64 * 64;  i += 256) wg[i] = Wg1[i];
    for (int i = t; i < 128 * 64; i += 256) wd[i] = Wd[i];
    for (int i = t; i < 64 * DD;  i += 256) wo[i] = Wo[i];
    if (t < 64) { sbg[t] = bg[t]; sbd[t] = bd[t]; }
    if (t < DD) sbo[t] = bo[t];
    __syncthreads();

    const int g = blockIdx.x * 256 + t;
    const int b = g >> 12;
    const int n = g & (NN - 1);
    const float alive = (colDeg[n] != 0) ? 1.0f : 0.0f;

    float av[64];
    for (int i = 0; i < 64; ++i) av[i] = agg[(size_t)n * BF + b * 64 + i];
    float G[64];
    for (int f = 0; f < 64; ++f) {
        float s = sbg[f];
        for (int gg = 0; gg < 64; ++gg) s += av[gg] * wg[gg * 64 + f];
        G[f] = tanhf(s) * alive;
    }
    float cv[64];
    for (int i = 0; i < 64; ++i) cv[i] = cond_f32[(size_t)n * BF + b * 64 + i];
    float h[64];
    for (int f = 0; f < 64; ++f) {
        float s = sbd[f];
        for (int c  = 0; c  < 64; ++c ) s += cv[c]  * wd[c * 64 + f];
        for (int gg = 0; gg < 64; ++gg) s += G[gg] * wd[(64 + gg) * 64 + f];
        h[f] = fmaxf(s, 0.0f);
    }
    const float4 xl = *(const float4*)(ts + (((size_t)b * TT + (TT - 1)) * NN + n) * DD);
    const float xlast[DD] = { xl.x, xl.y, xl.z, xl.w };
    #pragma unroll
    for (int d = 0; d < DD; ++d) {
        float s = sbo[d];
        for (int f = 0; f < 64; ++f) s += h[f] * wo[f * DD + d];
        out[((size_t)b * NN + n) * DD + d] = xlast[d] + tanhf(s);
    }
}

// ---------------------------------------------------------------------------
extern "C" void kernel_launch(void* const* d_in, const int* in_sizes, int n_in,
                              void* d_out, int out_size, void* d_ws, size_t ws_size,
                              hipStream_t stream)
{
    const float* ts    = (const float*)d_in[0];
    const int*   edges = (const int*)  d_in[1];
    const float* Wc1   = (const float*)d_in[2];
    const float* bc1   = (const float*)d_in[3];
    const float* Wc2   = (const float*)d_in[4];
    const float* bc2   = (const float*)d_in[5];
    const float* Wg1   = (const float*)d_in[6];
    const float* bg    = (const float*)d_in[7];
    const float* Wd    = (const float*)d_in[8];
    const float* bd    = (const float*)d_in[9];
    const float* Wo    = (const float*)d_in[10];
    const float* bo    = (const float*)d_in[11];
    float* out = (float*)d_out;

    // workspace layout (12 MB + 16 KB)
    char* ws = (char*)d_ws;
    float*    cond_f32 = (float*)   (ws);                        // 4 MB
    _Float16* cond_hi  = (_Float16*)(ws + 4  * 1024 * 1024);     // 2 MB
    _Float16* cond_lo  = (_Float16*)(ws + 6  * 1024 * 1024);     // 2 MB
    float*    agg      = (float*)   (ws + 8  * 1024 * 1024);     // 4 MB
    int*      colDeg   = (int*)     (ws + 12 * 1024 * 1024);     // 16 KB

    zero_kernel<<<NN / 256, 256, 0, stream>>>(colDeg, NN);
    cond_kernel<<<(BB * NN) / 256, 256, 0, stream>>>(ts, Wc1, bc1, Wc2, bc2,
                                                     cond_f32, cond_hi, cond_lo);
    agg_kernel<<<NN / MT, 256, 0, stream>>>(edges, cond_f32, cond_hi, cond_lo,
                                            agg, colDeg);
    out_kernel<<<(BB * NN) / 256, 256, 0, stream>>>(ts, cond_f32, agg, colDeg,
                                                    Wg1, bg, Wd, bd, Wo, bo, out);
}
